// STGraphTransformer_40200893890747
// MI455X (gfx1250) — matrix operand order repaired and checked by reading, not compiled, not run on hardware
//
#include <hip/hip_runtime.h>
#include <hip/hip_bf16.h>
#include <math.h>

// ---------------- problem constants (from reference) ----------------
#define NN   50000
#define EE   800000
#define HID  128
#define HEADS 4
#define DH   32
#define OUTC 10
#define EDIM 16

typedef __attribute__((ext_vector_type(16))) __bf16 v16bf;
typedef __attribute__((ext_vector_type(8)))  float  v8f;

// ---------------- helpers ----------------
__device__ __forceinline__ float wave_sum(float x) {
#pragma unroll
  for (int off = 16; off > 0; off >>= 1) x += __shfl_xor(x, off, 32);
  return x;
}
// order-preserving float<->uint encoding for atomicMax-based segment max
__device__ __forceinline__ unsigned fenc(float f) {
  unsigned u = __float_as_uint(f);
  return (u >> 31) ? ~u : (u | 0x80000000u);
}
__device__ __forceinline__ float fdec(unsigned u) {
  return __uint_as_float((u >> 31) ? (u & 0x7FFFFFFFu) : ~u);
}
// pack two f32 -> one dword holding two bf16 (lo = first)
__device__ __forceinline__ unsigned pkbf(float lo, float hi) {
  union { unsigned u; __bf16 h[2]; } t;
  t.h[0] = (__bf16)lo; t.h[1] = (__bf16)hi;
  return t.u;
}
union AFrag { v16bf v; unsigned u[8]; uint4 q[2]; };

// ---------------- WMMA GEMM: C[M,128] = A[M,KD] @ W[KD,128] (+ bias) ----------------
// 128 threads = 4 waves; WG computes 32 rows x 128 cols.
// wave w: row-tile rt=w&1, col-tiles (w>>1)+2j, j=0..3.
// LDS tiles are K-pair packed (one dword = bf16(k),bf16(k+1)), B transposed to
// fragment order [n][kpair] so every fragment gather is two ds_load_b128.
template <int KD>
__global__ __launch_bounds__(128) void stgt_lin128_wmma(
    const float* __restrict__ A, const float* __restrict__ W,
    const float* __restrict__ bias, float* __restrict__ C, int M) {
  __shared__ unsigned As2[32 * 16];     // [row][kpair]
  __shared__ unsigned Bs2T[128 * 16];   // [n][kpair]
  const int tid  = threadIdx.x;
  const int wave = tid >> 5, lane = tid & 31;
  const int half = lane >> 4, lrow = lane & 15, lcol = lane & 15;
  const int row0 = blockIdx.x * 32;
  const int rt = wave & 1, cg = wave >> 1;
  v8f acc[4] = {{}, {}, {}, {}};

  for (int k0 = 0; k0 < KD; k0 += 32) {
    // ---- stage A: 32 rows x 8 float4-quads = 256 tasks, 2/thread ----
#pragma unroll
    for (int i = 0; i < 2; ++i) {
      int tt = tid + i * 128;
      int r = tt >> 3, q4 = tt & 7;
      int gr = row0 + r, gk = k0 + q4 * 4;
      float4 v = make_float4(0.f, 0.f, 0.f, 0.f);
      if (gr < M && gk < KD)
        v = *reinterpret_cast<const float4*>(A + (size_t)gr * KD + gk);
      uint2 u; u.x = pkbf(v.x, v.y); u.y = pkbf(v.z, v.w);
      *reinterpret_cast<uint2*>(&As2[r * 16 + q4 * 2]) = u;
    }
    // ---- stage B: 512 tasks, kpair-fastest; scatter into [n][kpair] ----
#pragma unroll
    for (int i = 0; i < 4; ++i) {
      int tt = tid + i * 128;
      int kp = tt & 15, q = tt >> 4;          // q = n-quad 0..31
      int gk = k0 + 2 * kp;
      float4 lo = make_float4(0.f, 0.f, 0.f, 0.f), hi = lo;
      if (gk + 1 < KD) {
        lo = *reinterpret_cast<const float4*>(W + (size_t)gk * HID + q * 4);
        hi = *reinterpret_cast<const float4*>(W + (size_t)(gk + 1) * HID + q * 4);
      }
      Bs2T[(q * 4 + 0) * 16 + kp] = pkbf(lo.x, hi.x);
      Bs2T[(q * 4 + 1) * 16 + kp] = pkbf(lo.y, hi.y);
      Bs2T[(q * 4 + 2) * 16 + kp] = pkbf(lo.z, hi.z);
      Bs2T[(q * 4 + 3) * 16 + kp] = pkbf(lo.w, hi.w);
    }
    if (KD > 32 && k0 + 32 < KD)   // hint next A k-slice (global_prefetch_b8)
      __builtin_prefetch(A + (size_t)(row0 + (tid >> 2)) * KD + k0 + 32, 0, 0);
    __syncthreads();
    // ---- fragments: contiguous 2x ds_load_b128 each ----
    AFrag af;
    {
      int abase = (rt * 16 + lrow) * 16 + half * 4;
      af.q[0] = *reinterpret_cast<const uint4*>(&As2[abase]);
      af.q[1] = *reinterpret_cast<const uint4*>(&As2[abase + 8]);
    }
#pragma unroll
    for (int j = 0; j < 4; ++j) {
      int n0 = (cg + 2 * j) * 16;
      AFrag bf;
      int bbase = (n0 + lcol) * 16 + half * 4;
      bf.q[0] = *reinterpret_cast<const uint4*>(&Bs2T[bbase]);
      bf.q[1] = *reinterpret_cast<const uint4*>(&Bs2T[bbase + 8]);
      acc[j] = __builtin_amdgcn_wmma_f32_16x16x32_bf16(
          false, af.v, false, bf.v, (short)0, acc[j], false, false);
    }
    __syncthreads();
  }
  // ---- epilogue: D layout M = r + 8*half, N = lane&15 ----
#pragma unroll
  for (int j = 0; j < 4; ++j) {
    int n0 = (cg + 2 * j) * 16;
    float bv = bias ? bias[n0 + lcol] : 0.0f;
#pragma unroll
    for (int r = 0; r < 8; ++r) {
      int gr = row0 + rt * 16 + half * 8 + r;
      if (gr < M) C[(size_t)gr * HID + n0 + lcol] = acc[j][r] + bv;
    }
  }
}

// ---------------- conv1d (K=3, SAME) over node axis, 3 taps x WMMA ----------------
// H:[M,128]; CW:[o=128][i=128][t=3]; out = relu(conv + CB). 32-row tiles + 2 halo rows.
__global__ __launch_bounds__(128) void stgt_conv1d_wmma(
    const float* __restrict__ H, const float* __restrict__ CW,
    const float* __restrict__ CB, float* __restrict__ out, int M) {
  __shared__ unsigned As2[34 * 16];         // rows row0-1 .. row0+32, kpair-packed
  __shared__ unsigned Bs2cT[3 * 128 * 16];  // per-tap [n][kpair]
  const int tid  = threadIdx.x;
  const int wave = tid >> 5, lane = tid & 31;
  const int half = lane >> 4, lrow = lane & 15, lcol = lane & 15;
  const int row0 = blockIdx.x * 32;
  const int rt = wave & 1, cg = wave >> 1;
  v8f acc[4] = {{}, {}, {}, {}};

  for (int k0 = 0; k0 < HID; k0 += 32) {
    // ---- stage A: 34 rows x 8 quads = 272 tasks ----
#pragma unroll
    for (int i = 0; i < 3; ++i) {
      int tt = tid + i * 128;
      if (tt < 34 * 8) {
        int r = tt >> 3, q4 = tt & 7;
        int gr = row0 + r - 1;
        float4 v = make_float4(0.f, 0.f, 0.f, 0.f);
        if (gr >= 0 && gr < M)
          v = *reinterpret_cast<const float4*>(H + (size_t)gr * HID + k0 + q4 * 4);
        uint2 u; u.x = pkbf(v.x, v.y); u.y = pkbf(v.z, v.w);
        *reinterpret_cast<uint2*>(&As2[r * 16 + q4 * 2]) = u;
      }
    }
    // ---- stage B (all 3 taps): 2048 tasks, kpair-fastest (conflict-free stores,
    // 24B-stride coalesced CW reads). Pairs (k,t=0..2),(k+1,t=0..2) = 6 contiguous
    // floats -> 3 aligned float2 loads.
#pragma unroll
    for (int i = 0; i < 16; ++i) {
      int tt = tid + i * 128;
      int kp = tt & 15, n = tt >> 4;
      size_t base = (size_t)n * (HID * 3) + (size_t)(k0 + 2 * kp) * 3;
      float2 c01 = *reinterpret_cast<const float2*>(CW + base);
      float2 c23 = *reinterpret_cast<const float2*>(CW + base + 2);
      float2 c45 = *reinterpret_cast<const float2*>(CW + base + 4);
      Bs2cT[0 * 2048 + n * 16 + kp] = pkbf(c01.x, c23.y);  // tap0: (k),(k+1)
      Bs2cT[1 * 2048 + n * 16 + kp] = pkbf(c01.y, c45.x);  // tap1
      Bs2cT[2 * 2048 + n * 16 + kp] = pkbf(c23.x, c45.y);  // tap2
    }
    __syncthreads();
#pragma unroll
    for (int t = 0; t < 3; ++t) {
      AFrag af;
      {
        int abase = (rt * 16 + lrow + t) * 16 + half * 4;  // tap shift
        af.q[0] = *reinterpret_cast<const uint4*>(&As2[abase]);
        af.q[1] = *reinterpret_cast<const uint4*>(&As2[abase + 8]);
      }
#pragma unroll
      for (int j = 0; j < 4; ++j) {
        int n0 = (cg + 2 * j) * 16;
        AFrag bf;
        int bbase = t * 2048 + (n0 + lcol) * 16 + half * 4;
        bf.q[0] = *reinterpret_cast<const uint4*>(&Bs2cT[bbase]);
        bf.q[1] = *reinterpret_cast<const uint4*>(&Bs2cT[bbase + 8]);
        acc[j] = __builtin_amdgcn_wmma_f32_16x16x32_bf16(
            false, af.v, false, bf.v, (short)0, acc[j], false, false);
      }
    }
    __syncthreads();
  }
#pragma unroll
  for (int j = 0; j < 4; ++j) {
    int n0 = (cg + 2 * j) * 16;
    float bv = CB[n0 + lcol];
#pragma unroll
    for (int r = 0; r < 8; ++r) {
      int gr = row0 + rt * 16 + half * 8 + r;
      if (gr < M)
        out[(size_t)gr * HID + n0 + lcol] = fmaxf(acc[j][r] + bv, 0.0f);
    }
  }
}

// ---------------- attention pass A: alpha + segment max (wave per edge) ----------------
__global__ __launch_bounds__(256) void stgt_attn_alpha(
    const long long* __restrict__ ei, const float* __restrict__ q,
    const float* __restrict__ k, const float* __restrict__ ebuf,
    float* __restrict__ alpha, unsigned* __restrict__ amax) {
  int widx = (int)((blockIdx.x * (unsigned)blockDim.x + threadIdx.x) >> 5);
  int lane = threadIdx.x & 31;
  if (widx >= EE) return;
  int s = (int)ei[widx];
  int d = (int)ei[EE + widx];
  const float scale = 0.1767766952966369f;  // 1/sqrt(32)
#pragma unroll
  for (int h = 0; h < HEADS; ++h) {
    int c = h * DH + lane;
    float kj = k[(size_t)s * HID + c] + ebuf[(size_t)widx * HID + c];
    float p  = q[(size_t)d * HID + c] * kj;
    p = wave_sum(p) * scale;
    if (lane == h) {
      alpha[(size_t)widx * HEADS + h] = p;
      atomicMax(&amax[(size_t)d * HEADS + h], fenc(p));
    }
  }
}

// ---------------- attention pass B: exp + segment sum (thread per edge*head) -----------
__global__ __launch_bounds__(256) void stgt_attn_expsum(
    const long long* __restrict__ ei, const float* __restrict__ alpha,
    const unsigned* __restrict__ amax, float* __restrict__ ex,
    float* __restrict__ denom) {
  long long idx = (long long)blockIdx.x * blockDim.x + threadIdx.x;
  if (idx >= (long long)EE * HEADS) return;
  int e = (int)(idx >> 2), h = (int)(idx & 3);
  int d = (int)ei[EE + e];
  float m = fdec(amax[(size_t)d * HEADS + h]);
  float v = __expf(alpha[idx] - m);
  ex[idx] = v;
  atomicAdd(&denom[(size_t)d * HEADS + h], v);
}

// ---------------- attention pass C: weighted scatter (wave per edge) -------------------
__global__ __launch_bounds__(256) void stgt_attn_scatter(
    const long long* __restrict__ ei, const float* __restrict__ v,
    const float* __restrict__ ebuf, const float* __restrict__ ex,
    const float* __restrict__ denom, float* __restrict__ outacc) {
  int widx = (int)((blockIdx.x * (unsigned)blockDim.x + threadIdx.x) >> 5);
  int lane = threadIdx.x & 31;
  if (widx >= EE) return;
  int s = (int)ei[widx];
  int d = (int)ei[EE + widx];
#pragma unroll
  for (int h = 0; h < HEADS; ++h) {
    int c = h * DH + lane;
    float vj = v[(size_t)s * HID + c] + ebuf[(size_t)widx * HID + c];
    float a  = ex[(size_t)widx * HEADS + h] /
               (denom[(size_t)d * HEADS + h] + 1e-16f);
    atomicAdd(&outacc[(size_t)d * HID + c], vj * a);
  }
}

// ---------------- combine + layernorm + exact GELU (wave per node) ---------------------
__global__ __launch_bounds__(256) void stgt_combine_ln_gelu(
    const float* __restrict__ attnout, const float* __restrict__ skip,
    const float* __restrict__ resid, const float* __restrict__ g,
    const float* __restrict__ b, float* __restrict__ hout, int M) {
  int n = (int)((blockIdx.x * (unsigned)blockDim.x + threadIdx.x) >> 5);
  int lane = threadIdx.x & 31;
  if (n >= M) return;
  float vals[4];
  float s = 0.0f;
#pragma unroll
  for (int i = 0; i < 4; ++i) {
    size_t idx = (size_t)n * HID + i * 32 + lane;
    vals[i] = attnout[idx] + skip[idx] + resid[idx];
    s += vals[i];
  }
  float mu = wave_sum(s) * (1.0f / HID);
  float var = 0.0f;
#pragma unroll
  for (int i = 0; i < 4; ++i) { float dd = vals[i] - mu; var += dd * dd; }
  var = wave_sum(var) * (1.0f / HID);
  float rstd = rsqrtf(var + 1e-5f);
#pragma unroll
  for (int i = 0; i < 4; ++i) {
    int c = i * 32 + lane;
    float y = (vals[i] - mu) * rstd * g[c] + b[c];
    float ge = 0.5f * y * (1.0f + erff(y * 0.7071067811865475f));
    hout[(size_t)n * HID + c] = ge;
  }
}

// ---------------- output linear (128->10) + log_softmax (wave per node) ----------------
__global__ __launch_bounds__(256) void stgt_out_logsoftmax(
    const float* __restrict__ H, const float* __restrict__ Wo,
    const float* __restrict__ bo, float* __restrict__ out, int M) {
  int n = (int)((blockIdx.x * (unsigned)blockDim.x + threadIdx.x) >> 5);
  int lane = threadIdx.x & 31;
  if (n >= M) return;
  float hv[4];
#pragma unroll
  for (int i = 0; i < 4; ++i) hv[i] = H[(size_t)n * HID + i * 32 + lane];
  float lg[OUTC];
#pragma unroll
  for (int o = 0; o < OUTC; ++o) {
    float p = 0.0f;
#pragma unroll
    for (int i = 0; i < 4; ++i) {
      int c = i * 32 + lane;
      p += hv[i] * Wo[(size_t)c * OUTC + o];
    }
    lg[o] = wave_sum(p) + bo[o];
  }
  float m = lg[0];
#pragma unroll
  for (int o = 1; o < OUTC; ++o) m = fmaxf(m, lg[o]);
  float se = 0.0f;
#pragma unroll
  for (int o = 0; o < OUTC; ++o) se += __expf(lg[o] - m);
  float lse = m + __logf(se);
  if (lane == 0) {
#pragma unroll
    for (int o = 0; o < OUTC; ++o) out[(size_t)n * OUTC + o] = lg[o] - lse;
  }
}

// ---------------- host orchestration ----------------
extern "C" void kernel_launch(void* const* d_in, const int* in_sizes, int n_in,
                              void* d_out, int out_size, void* d_ws, size_t ws_size,
                              hipStream_t stream) {
  // setup_inputs() dict order: x, edge_index, edge_attr, params{W_in,b_in,W_out,b_out,
  //   layers[3]{conv_w,conv_b,Wq,bq,Wk,bk,Wv,bv,We,Ws,bs,ln_g,ln_b}}
  const float*     x     = (const float*)d_in[0];
  const long long* ei    = (const long long*)d_in[1];
  const float*     ea    = (const float*)d_in[2];
  const float*     W_in  = (const float*)d_in[3];
  const float*     b_in  = (const float*)d_in[4];
  const float*     W_out = (const float*)d_in[5];
  const float*     b_out = (const float*)d_in[6];
  struct Layer {
    const float *cw, *cb, *Wq, *bq, *Wk, *bk, *Wv, *bv, *We, *Ws, *bs, *lng, *lnb;
  } L[3];
  int p = 7;
  for (int l = 0; l < 3; ++l) {
    L[l].cw = (const float*)d_in[p++]; L[l].cb = (const float*)d_in[p++];
    L[l].Wq = (const float*)d_in[p++]; L[l].bq = (const float*)d_in[p++];
    L[l].Wk = (const float*)d_in[p++]; L[l].bk = (const float*)d_in[p++];
    L[l].Wv = (const float*)d_in[p++]; L[l].bv = (const float*)d_in[p++];
    L[l].We = (const float*)d_in[p++]; L[l].Ws = (const float*)d_in[p++];
    L[l].bs = (const float*)d_in[p++]; L[l].lng = (const float*)d_in[p++];
    L[l].lnb = (const float*)d_in[p++];
  }

  // workspace layout (f32 words)
  const size_t NH = (size_t)NN * HID;
  float* ws      = (float*)d_ws;
  float* h       = ws;                 // [N,128]
  float* hc      = h + NH;             // [N,128] conv output / residual
  float* q       = hc + NH;
  float* kbuf    = q + NH;
  float* v       = kbuf + NH;
  float* skip    = v + NH;
  float* attnout = skip + NH;          // [N,128] accumulator
  float* ebuf    = attnout + NH;       // [E,128] edge projection
  float* alpha   = ebuf + (size_t)EE * HID;      // [E,4]
  float* exbuf   = alpha + (size_t)EE * HEADS;   // [E,4]
  unsigned* amax = (unsigned*)(exbuf + (size_t)EE * HEADS);  // [N,4]
  float* denom   = (float*)(amax + (size_t)NN * HEADS);      // [N,4]

  const int gemmN_blocks = (NN + 31) / 32;
  const int gemmE_blocks = (EE + 31) / 32;
  const int waveN_blocks = (NN + 7) / 8;   // 256 thr = 8 waves
  const int waveE_blocks = (EE + 7) / 8;
  const int ehBlocks     = (int)(((long long)EE * HEADS + 255) / 256);

  // h = x @ W_in + b_in
  stgt_lin128_wmma<32><<<gemmN_blocks, 128, 0, stream>>>(x, W_in, b_in, h, NN);

  for (int l = 0; l < 3; ++l) {
    // temporal conv + relu -> hc (also the residual)
    stgt_conv1d_wmma<<<gemmN_blocks, 128, 0, stream>>>(h, L[l].cw, L[l].cb, hc, NN);
    // q/k/v/skip projections from hc
    stgt_lin128_wmma<128><<<gemmN_blocks, 128, 0, stream>>>(hc, L[l].Wq, L[l].bq, q, NN);
    stgt_lin128_wmma<128><<<gemmN_blocks, 128, 0, stream>>>(hc, L[l].Wk, L[l].bk, kbuf, NN);
    stgt_lin128_wmma<128><<<gemmN_blocks, 128, 0, stream>>>(hc, L[l].Wv, L[l].bv, v, NN);
    stgt_lin128_wmma<128><<<gemmN_blocks, 128, 0, stream>>>(hc, L[l].Ws, L[l].bs, skip, NN);
    // edge projection: ebuf = edge_attr @ We  (K=16 zero-padded to 32)
    stgt_lin128_wmma<EDIM><<<gemmE_blocks, 128, 0, stream>>>(ea, L[l].We, nullptr, ebuf, EE);
    // zero the per-layer accumulators (deterministic across graph replays)
    hipMemsetAsync(attnout, 0, NH * sizeof(float), stream);
    hipMemsetAsync(amax, 0, (size_t)NN * HEADS * (sizeof(unsigned) + sizeof(float)), stream);
    // attention
    stgt_attn_alpha<<<waveE_blocks, 256, 0, stream>>>(ei, q, kbuf, ebuf, alpha, amax);
    stgt_attn_expsum<<<ehBlocks, 256, 0, stream>>>(ei, alpha, amax, exbuf, denom);
    stgt_attn_scatter<<<waveE_blocks, 256, 0, stream>>>(ei, v, ebuf, exbuf, denom, attnout);
    // combine + LN + GELU -> h (next layer input)
    stgt_combine_ln_gelu<<<waveN_blocks, 256, 0, stream>>>(attnout, skip, hc,
                                                           L[l].lng, L[l].lnb, h, NN);
  }
  // logits + log_softmax
  stgt_out_logsoftmax<<<waveN_blocks, 256, 0, stream>>>(h, W_out, b_out, (float*)d_out, NN);
}
